// Attention3D_5738076308102
// MI455X (gfx1250) — matrix operand order repaired and checked by reading, not compiled, not run on hardware
//
#include <hip/hip_runtime.h>
#include <hip/hip_bf16.h>

// ---------------------------------------------------------------------------
// Fused 3D attention block for MI455X (gfx1250), wave32, WMMA bf16 path.
// B=2, C=128, D=H=W=16 -> N=4096, NH=4, HD=32, GROUPS=32.
// ---------------------------------------------------------------------------

typedef __bf16 bf16;
typedef __attribute__((ext_vector_type(16))) __bf16 bf16x16;
typedef __attribute__((ext_vector_type(8)))  __bf16 bf16x8;
typedef __attribute__((ext_vector_type(8)))  float  f32x8;

union ABFrag { bf16x16 v; bf16x8 h[2]; };

#define NB      2
#define NC      128
#define NN      4096
#define NHEADS  4
#define HDIM    32
#define SCALE_Q 0.17677669529663687f  // 32^-0.5

static __device__ __forceinline__ f32x8 wmma_bf16(const ABFrag& a, const ABFrag& b, f32x8 c) {
    return __builtin_amdgcn_wmma_f32_16x16x32_bf16(false, a.v, false, b.v, (short)0, c,
                                                   false, false);
}

// --------------------------- kernel 0: weight convert ----------------------
__global__ __launch_bounds__(256) void k_wconv(const float* __restrict__ qkvw,
                                               const float* __restrict__ outw,
                                               bf16* __restrict__ wq, bf16* __restrict__ wo) {
    int i = blockIdx.x * 256 + threadIdx.x;
    if (i < 3 * NC * NC) wq[i] = (bf16)qkvw[i];
    if (i < NC * NC)     wo[i] = (bf16)outw[i];
}

// --------------------------- kernel 1: group norm --------------------------
// block per (b, group): 4 channels x 4096 spatial = 16384 elems.
// writes xn transposed: xnt[b][n][c] (bf16, c contiguous).
__global__ __launch_bounds__(256) void k_gnorm(const float* __restrict__ x,
                                               const float* __restrict__ gw,
                                               const float* __restrict__ gb,
                                               bf16* __restrict__ xnt) {
    const int b = blockIdx.x >> 5;
    const int g = blockIdx.x & 31;
    const int tid = threadIdx.x;
    const float* xg = x + ((size_t)b * NC + g * 4) * NN;

    float s = 0.f, ss = 0.f;
    for (int i = tid; i < 4 * NN; i += 256) { float v = xg[i]; s += v; ss += v * v; }

    __shared__ float rs[256], rss[256];
    __shared__ float mu_s, rstd_s;
    rs[tid] = s; rss[tid] = ss;
    __syncthreads();
    for (int st = 128; st > 0; st >>= 1) {
        if (tid < st) { rs[tid] += rs[tid + st]; rss[tid] += rss[tid + st]; }
        __syncthreads();
    }
    if (tid == 0) {
        float mu = rs[0] * (1.f / 16384.f);
        float var = rss[0] * (1.f / 16384.f) - mu * mu;
        mu_s = mu; rstd_s = rsqrtf(var + 1e-5f);
    }
    __syncthreads();
    const float mu = mu_s, rstd = rstd_s;

    float wc[4], bc[4];
    #pragma unroll
    for (int c = 0; c < 4; ++c) {
        float w = gw[g * 4 + c] * rstd;
        wc[c] = w;
        bc[c] = gb[g * 4 + c] - mu * w;
    }
    for (int n = tid; n < NN; n += 256) {
        union { bf16 e[4]; uint2 u; } pk;
        #pragma unroll
        for (int c = 0; c < 4; ++c) pk.e[c] = (bf16)(xg[c * NN + n] * wc[c] + bc[c]);
        *(uint2*)&xnt[((size_t)b * NN + n) * NC + g * 4] = pk.u;
    }
}

// --------------------------- kernel 2: QKV GEMM ----------------------------
// qkv[o,n] = sum_c w[o,c]*xn[c,n] + bias.  A = w (16x32 tiles), B = xnt.
// Each wave: one 16(M) x 64(N) tile.  Waves = B * 24 * 64 = 3072 -> 384 blocks.
__global__ __launch_bounds__(256) void k_qkv(const bf16* __restrict__ wq,
                                             const float* __restrict__ qkv_b,
                                             const bf16* __restrict__ xnt,
                                             bf16* __restrict__ qt,
                                             bf16* __restrict__ kt,
                                             bf16* __restrict__ vv) {
    const int lane = threadIdx.x & 31;
    const int wid  = blockIdx.x * 8 + (threadIdx.x >> 5);
    const int ng = wid & 63;
    const int mt = (wid >> 6) % 24;
    const int b  = wid / (64 * 24);
    const int lm = lane & 15, half = lane >> 4;
    const int o0 = mt * 16;
    const int n0 = ng * 64;

    f32x8 acc[4] = {{}, {}, {}, {}};
    const bf16* wrow = wq + (size_t)(o0 + lm) * NC;
    #pragma unroll
    for (int kk = 0; kk < NC; kk += 32) {
        ABFrag a;
        a.h[0] = *(const bf16x8*)(wrow + kk + 16 * half);
        a.h[1] = *(const bf16x8*)(wrow + kk + 16 * half + 8);
        #pragma unroll
        for (int t = 0; t < 4; ++t) {
            const bf16* brow = xnt + ((size_t)b * NN + (n0 + t * 16 + lm)) * NC + kk + 16 * half;
            ABFrag bb;
            bb.h[0] = *(const bf16x8*)(brow);
            bb.h[1] = *(const bf16x8*)(brow + 8);
            acc[t] = wmma_bf16(a, bb, acc[t]);
        }
    }

    float bias[8];
    #pragma unroll
    for (int j = 0; j < 8; ++j) bias[j] = qkv_b[o0 + 8 * half + j];
    const int which = o0 >> 7;                 // 0=q, 1=k, 2=v
    const int hh    = (o0 & 127) >> 5;
    const int dbase = (o0 & 31) + 8 * half;

    #pragma unroll
    for (int t = 0; t < 4; ++t) {
        const int n = n0 + t * 16 + lm;
        if (which == 2) {
            // v stored [b][h][d][n]
            #pragma unroll
            for (int j = 0; j < 8; ++j)
                vv[(((size_t)b * NHEADS + hh) * HDIM + dbase + j) * NN + n] =
                    (bf16)(acc[t][j] + bias[j]);
        } else {
            const float sc = (which == 0) ? SCALE_Q : 1.0f;
            union { bf16x8 v; bf16 e[8]; } pk;
            #pragma unroll
            for (int j = 0; j < 8; ++j) pk.e[j] = (bf16)((acc[t][j] + bias[j]) * sc);
            bf16* dst = (which == 0 ? qt : kt) +
                        (((size_t)b * NHEADS + hh) * NN + n) * HDIM + dbase;
            *(bf16x8*)dst = pk.v;
        }
    }
}

// --------------------------- kernel 3: flash attention ---------------------
// S^T[m,n] = sum_d k[m,d] q[n,d]  (A = k_t tile, B = q_t frag, K=32)
// O[d,n]  += sum_m v[d,m] P^T[m,n] (A = v tile, B = P^T from LDS, K=32)
// One wave per 32-query group (two 16-col B frags, two softmax states);
// k/v chunk fragments double-buffered across iterations.
// Waves = B*NH*(N/32) = 1024 -> 128 blocks.
__global__ __launch_bounds__(256) void k_attn(const bf16* __restrict__ qt,
                                              const bf16* __restrict__ kt,
                                              const bf16* __restrict__ vv,
                                              bf16* __restrict__ ot) {
    __shared__ alignas(16) bf16 lds[8][2][16][32];
    const int lane  = threadIdx.x & 31;
    const int wslot = threadIdx.x >> 5;
    const int wid   = blockIdx.x * 8 + wslot;
    const int qg = wid & 127;                  // 32-query group
    const int bh = wid >> 7;                   // b*NHEADS + h
    const int lm = lane & 15, half = lane >> 4;
    const int n0 = qg * 32;

    // B fragments: q for this wave's 32 queries (reused every chunk)
    ABFrag bq0, bq1;
    {
        const bf16* qp = qt + ((size_t)bh * NN + n0 + lm) * HDIM + 8 * half;
        bq0.h[0] = *(const bf16x8*)qp;
        bq0.h[1] = *(const bf16x8*)(qp + 16);
        const bf16* qp1 = qp + 16 * HDIM;
        bq1.h[0] = *(const bf16x8*)qp1;
        bq1.h[1] = *(const bf16x8*)(qp1 + 16);
    }

    const bf16* kbase = kt + (size_t)bh * NN * HDIM;
    const bf16* vbase = vv + (size_t)bh * HDIM * NN;

    f32x8 oa00 = {}, oa01 = {}, oa10 = {}, oa11 = {};   // [dtile][qgroup]
    float Mn0 = -1e30f, ln0 = 0.f, Mn1 = -1e30f, ln1 = 0.f;

    // ---- preload chunk 0 fragments (double buffer)
    ABFrag ak0, ak1, av0, av1;
    {
        const bf16* kp = kbase + (size_t)lm * HDIM + 8 * half;
        ak0.h[0] = *(const bf16x8*)kp;        ak0.h[1] = *(const bf16x8*)(kp + 16);
        ak1.h[0] = *(const bf16x8*)(kp + 512); ak1.h[1] = *(const bf16x8*)(kp + 528);
        const bf16* vp = vbase + (size_t)lm * NN + 8 * half;
        av0.h[0] = *(const bf16x8*)vp;        av0.h[1] = *(const bf16x8*)(vp + 16);
        const bf16* vp1 = vp + 16 * NN;
        av1.h[0] = *(const bf16x8*)vp1;       av1.h[1] = *(const bf16x8*)(vp1 + 16);
    }

    for (int m0 = 0; m0 < NN; m0 += 32) {
        // ---- S^T tiles: 4 WMMAs (2 m-tiles x 2 q-groups) over K=32 head dims
        f32x8 z = {};
        f32x8 s00 = wmma_bf16(ak0, bq0, z);    // m rows m0+8h+j,    q cols n0+lm
        f32x8 s10 = wmma_bf16(ak1, bq0, z);    // m rows m0+16+8h+j
        f32x8 s01 = wmma_bf16(ak0, bq1, z);    //                    q cols n0+16+lm
        f32x8 s11 = wmma_bf16(ak1, bq1, z);

        // ---- issue next chunk's k/v loads (consumed next iteration)
        ABFrag nk0, nk1, nv0, nv1;
        {
            const int mnext = (m0 + 32 < NN) ? m0 + 32 : m0;
            const bf16* kp = kbase + (size_t)(mnext + lm) * HDIM + 8 * half;
            nk0.h[0] = *(const bf16x8*)kp;         nk0.h[1] = *(const bf16x8*)(kp + 16);
            nk1.h[0] = *(const bf16x8*)(kp + 512); nk1.h[1] = *(const bf16x8*)(kp + 528);
            const bf16* vp = vbase + (size_t)lm * NN + mnext + 8 * half;
            nv0.h[0] = *(const bf16x8*)vp;         nv0.h[1] = *(const bf16x8*)(vp + 16);
            const bf16* vp1 = vp + 16 * NN;
            nv1.h[0] = *(const bf16x8*)vp1;        nv1.h[1] = *(const bf16x8*)(vp1 + 16);
        }

        // ---- online softmax, query group 0 (lane's column n = n0+lm)
        union { bf16x8 v; bf16 e[8]; } p00, p10, p01, p11;
        {
            float tm = s00[0];
            #pragma unroll
            for (int j = 0; j < 8; ++j) { tm = fmaxf(tm, s00[j]); tm = fmaxf(tm, s10[j]); }
            tm = fmaxf(tm, __shfl_xor(tm, 16, 32));
            const float Mnew  = fmaxf(Mn0, tm);
            const float alpha = __expf(Mn0 - Mnew);
            float ts = 0.f;
            #pragma unroll
            for (int j = 0; j < 8; ++j) {
                float e0 = __expf(s00[j] - Mnew);
                float e1 = __expf(s10[j] - Mnew);
                ts += e0 + e1;
                p00.e[j] = (bf16)e0;
                p10.e[j] = (bf16)e1;
            }
            ts += __shfl_xor(ts, 16, 32);
            ln0 = ln0 * alpha + ts;
            Mn0 = Mnew;
            #pragma unroll
            for (int j = 0; j < 8; ++j) { oa00[j] *= alpha; oa10[j] *= alpha; }
        }
        // ---- online softmax, query group 1
        {
            float tm = s01[0];
            #pragma unroll
            for (int j = 0; j < 8; ++j) { tm = fmaxf(tm, s01[j]); tm = fmaxf(tm, s11[j]); }
            tm = fmaxf(tm, __shfl_xor(tm, 16, 32));
            const float Mnew  = fmaxf(Mn1, tm);
            const float alpha = __expf(Mn1 - Mnew);
            float ts = 0.f;
            #pragma unroll
            for (int j = 0; j < 8; ++j) {
                float e0 = __expf(s01[j] - Mnew);
                float e1 = __expf(s11[j] - Mnew);
                ts += e0 + e1;
                p01.e[j] = (bf16)e0;
                p11.e[j] = (bf16)e1;
            }
            ts += __shfl_xor(ts, 16, 32);
            ln1 = ln1 * alpha + ts;
            Mn1 = Mnew;
            #pragma unroll
            for (int j = 0; j < 8; ++j) { oa01[j] *= alpha; oa11[j] *= alpha; }
        }

        // ---- P^T -> LDS (per-wave tiles, rows m contiguous per lane)
        *(bf16x8*)&lds[wslot][0][lm][8 * half]      = p00.v;
        *(bf16x8*)&lds[wslot][0][lm][16 + 8 * half] = p10.v;
        *(bf16x8*)&lds[wslot][1][lm][8 * half]      = p01.v;
        *(bf16x8*)&lds[wslot][1][lm][16 + 8 * half] = p11.v;
        asm volatile("s_wait_dscnt 0" ::: "memory");

        ABFrag bp0, bp1;  // B fragments of P^T (K=m chunk of 32, N=n tile)
        bp0.h[0] = *(const bf16x8*)&lds[wslot][0][lm][8 * half];
        bp0.h[1] = *(const bf16x8*)&lds[wslot][0][lm][8 * half + 16];
        bp1.h[0] = *(const bf16x8*)&lds[wslot][1][lm][8 * half];
        bp1.h[1] = *(const bf16x8*)&lds[wslot][1][lm][8 * half + 16];

        // ---- O accumulation: A = v tiles (d rows), B = P^T (4 WMMAs)
        oa00 = wmma_bf16(av0, bp0, oa00);
        oa10 = wmma_bf16(av1, bp0, oa10);
        oa01 = wmma_bf16(av0, bp1, oa01);
        oa11 = wmma_bf16(av1, bp1, oa11);

        // ---- rotate double buffer
        ak0 = nk0; ak1 = nk1; av0 = nv0; av1 = nv1;
    }

    // ---- finalize: O /= l, write ot[b][n][c] (c = h*32 + d, contiguous per lane)
    const float rl0 = 1.f / ln0, rl1 = 1.f / ln1;
    const int b = bh >> 2, h = bh & 3;
    union { bf16x8 v; bf16 e[8]; } w00, w10, w01, w11;
    #pragma unroll
    for (int j = 0; j < 8; ++j) {
        w00.e[j] = (bf16)(oa00[j] * rl0);
        w10.e[j] = (bf16)(oa10[j] * rl0);
        w01.e[j] = (bf16)(oa01[j] * rl1);
        w11.e[j] = (bf16)(oa11[j] * rl1);
    }
    bf16* orow0 = ot + ((size_t)b * NN + n0 + lm) * NC + h * HDIM + 8 * half;
    *(bf16x8*)orow0        = w00.v;
    *(bf16x8*)(orow0 + 16) = w10.v;
    bf16* orow1 = orow0 + 16 * NC;
    *(bf16x8*)orow1        = w01.v;
    *(bf16x8*)(orow1 + 16) = w11.v;
}

// --------------------------- kernel 4: out proj + residual -----------------
// y[o,n] = sum_c ow[o,c]*o[c,n] + out_b[o] + x[o,n].  1024 waves -> 128 blocks.
__global__ __launch_bounds__(256) void k_oproj(const bf16* __restrict__ wo,
                                               const float* __restrict__ ob,
                                               const bf16* __restrict__ ot,
                                               const float* __restrict__ x,
                                               float* __restrict__ y) {
    const int lane = threadIdx.x & 31;
    const int wid  = blockIdx.x * 8 + (threadIdx.x >> 5);
    const int ng = wid & 63;
    const int mt = (wid >> 6) & 7;
    const int b  = wid >> 9;
    const int lm = lane & 15, half = lane >> 4;
    const int o0 = mt * 16;
    const int n0 = ng * 64;

    f32x8 acc[4] = {{}, {}, {}, {}};
    const bf16* wrow = wo + (size_t)(o0 + lm) * NC;
    #pragma unroll
    for (int kk = 0; kk < NC; kk += 32) {
        ABFrag a;
        a.h[0] = *(const bf16x8*)(wrow + kk + 16 * half);
        a.h[1] = *(const bf16x8*)(wrow + kk + 16 * half + 8);
        #pragma unroll
        for (int t = 0; t < 4; ++t) {
            const bf16* brow = ot + ((size_t)b * NN + (n0 + t * 16 + lm)) * NC + kk + 16 * half;
            ABFrag bb;
            bb.h[0] = *(const bf16x8*)(brow);
            bb.h[1] = *(const bf16x8*)(brow + 8);
            acc[t] = wmma_bf16(a, bb, acc[t]);
        }
    }
    float bias[8];
    #pragma unroll
    for (int j = 0; j < 8; ++j) bias[j] = ob[o0 + 8 * half + j];
    #pragma unroll
    for (int t = 0; t < 4; ++t) {
        const int n = n0 + t * 16 + lm;
        #pragma unroll
        for (int j = 0; j < 8; ++j) {
            const size_t idx = ((size_t)b * NC + o0 + 8 * half + j) * NN + n;
            y[idx] = acc[t][j] + bias[j] + x[idx];
        }
    }
}

// --------------------------- host launcher ---------------------------------
extern "C" void kernel_launch(void* const* d_in, const int* in_sizes, int n_in,
                              void* d_out, int out_size, void* d_ws, size_t ws_size,
                              hipStream_t stream) {
    (void)in_sizes; (void)n_in; (void)out_size; (void)ws_size;
    const float* x     = (const float*)d_in[0];
    const float* gn_w  = (const float*)d_in[1];
    const float* gn_b  = (const float*)d_in[2];
    const float* qkv_w = (const float*)d_in[3];
    const float* qkv_b = (const float*)d_in[4];
    const float* out_w = (const float*)d_in[5];
    const float* out_b = (const float*)d_in[6];
    float* y = (float*)d_out;

    // workspace layout (bytes); total ~10.2 MB
    char* ws = (char*)d_ws;
    bf16* wq_bf  = (bf16*)(ws + 0);                       // 384*128        (98304 B)
    bf16* wo_bf  = (bf16*)(ws + 98304);                   // 128*128        (32768 B)
    bf16* xnt    = (bf16*)(ws + 131072);                  // B*N*C          (2 MiB)
    bf16* qt     = (bf16*)(ws + 131072 + 2097152);        // B*NH*N*HD      (2 MiB)
    bf16* kt     = (bf16*)(ws + 131072 + 2 * 2097152);    // B*NH*N*HD      (2 MiB)
    bf16* vv     = (bf16*)(ws + 131072 + 3 * 2097152);    // B*NH*HD*N      (2 MiB)
    bf16* ot     = (bf16*)(ws + 131072 + 4 * 2097152);    // B*N*C          (2 MiB)

    k_wconv<<<192, 256, 0, stream>>>(qkv_w, out_w, wq_bf, wo_bf);
    k_gnorm<<<NB * 32, 256, 0, stream>>>(x, gn_w, gn_b, xnt);
    k_qkv  <<<384, 256, 0, stream>>>(wq_bf, qkv_b, xnt, qt, kt, vv);
    k_attn <<<128, 256, 0, stream>>>(qt, kt, vv, ot);
    k_oproj<<<128, 256, 0, stream>>>(wo_bf, out_b, ot, x, y);
}
